// MultiHeadedAttention_53626961658052
// MI455X (gfx1250) — matrix operand order repaired and checked by reading, not compiled, not run on hardware
//
#include <hip/hip_runtime.h>

#define B_  4
#define S_  2048
#define DM  1024
#define H_  16
#define DK  64
#define DV  64

typedef __attribute__((ext_vector_type(16))) __bf16 v16bf;
typedef __attribute__((ext_vector_type(8)))  __bf16 v8bf;
typedef __attribute__((ext_vector_type(8)))  float  v8f;
typedef __attribute__((ext_vector_type(4)))  float  v4f;

__device__ __forceinline__ v16bf join16(v8bf lo, v8bf hi8) {
  return __builtin_shufflevector(lo, hi8, 0, 1, 2, 3, 4, 5, 6, 7,
                                          8, 9, 10, 11, 12, 13, 14, 15);
}

// Async DMA 32 bytes/lane global -> LDS (two b128 beats), ASYNCcnt-tracked.
__device__ __forceinline__ void async_g2l_32B(__bf16* lds_dst,
                                              const __bf16* gsrc) {
  unsigned lo = (unsigned)(size_t)lds_dst;
  unsigned long long ga = (unsigned long long)(size_t)gsrc;
  asm volatile("global_load_async_to_lds_b128 %0, %1, off"
               :: "v"(lo), "v"(ga) : "memory");
  asm volatile("global_load_async_to_lds_b128 %0, %1, off offset:16"
               :: "v"(lo), "v"(ga) : "memory");
}

__device__ __forceinline__ void wait_async0() {
  asm volatile("s_wait_asynccnt 0x0" ::: "memory");
}

// ---------------------------------------------------------------------------
// Kernel 0a: generic f32 -> bf16 conversion, 8 elements / thread.
// ---------------------------------------------------------------------------
__global__ void cvt_kernel(const float* __restrict__ src,
                           __bf16* __restrict__ dst, int n8) {
  int i = blockIdx.x * 256 + threadIdx.x;
  if (i < n8) {
    v4f a = *(const v4f*)(src + (size_t)i * 8);
    v4f b = *(const v4f*)(src + (size_t)i * 8 + 4);
    v8bf o;
#pragma unroll
    for (int e = 0; e < 4; ++e) { o[e] = (__bf16)a[e]; o[4 + e] = (__bf16)b[e]; }
    *(v8bf*)(dst + (size_t)i * 8) = o;
  }
}

// ---------------------------------------------------------------------------
// Kernel 0b: zero the mean-output region + convert value -> bf16 ws.
// ---------------------------------------------------------------------------
__global__ void prep_kernel(const float* __restrict__ value,
                            float* __restrict__ out_mean,
                            __bf16* __restrict__ Vbf, int n8) {
  int i = blockIdx.x * 256 + threadIdx.x;
  if (i < n8) {
    v4f z = {};
    *(v4f*)(out_mean + (size_t)i * 8)     = z;
    *(v4f*)(out_mean + (size_t)i * 8 + 4) = z;
    v4f a = *(const v4f*)(value + (size_t)i * 8);
    v4f b = *(const v4f*)(value + (size_t)i * 8 + 4);
    v8bf o;
#pragma unroll
    for (int e = 0; e < 4; ++e) { o[e] = (__bf16)a[e]; o[4 + e] = (__bf16)b[e]; }
    *(v8bf*)(Vbf + (size_t)i * 8) = o;
  }
}

// ---------------------------------------------------------------------------
// Kernel 1: projection GEMM.  out[m,n] = sum_k X[m,k]*W[n,k] + bias[n]
// Block = 4 waves, 64(M) x 128(N); wave = 16(M) x 128(N), K step 32.
// The shared 128-col W tile is double-buffered in LDS via
// global_load_async_to_lds_b128 (1 row / thread / buffer).
// ---------------------------------------------------------------------------
__global__ __launch_bounds__(128) void proj_kernel(
    const float*  __restrict__ X,      // [B*S, DM] row-major f32
    const __bf16* __restrict__ Wbf,    // [DM, DM]  row-major bf16 (k contig)
    const float*  __restrict__ bias,   // [DM]
    float*        __restrict__ q_out,  // head-major f32 out or nullptr
    __bf16*       __restrict__ bfout,  // head-major bf16 ws
    float bf_scale)
{
  __shared__ __bf16 wtile[2 * 128 * 32];   // [buf][row][k]  16 KB

  const int tid  = threadIdx.x;
  const int wave = tid >> 5;
  const int lane = tid & 31;
  const int hi   = lane >> 4;
  const int ln   = lane & 15;

  const int tileM = (blockIdx.x * 4 + wave) * 16;
  const int tileN = blockIdx.y * 128;

  // DMA share: thread t owns W row (tileN + t), 64 B per K step.
  const __bf16* wsrc0 = Wbf + (size_t)(tileN + tid) * DM;
  __bf16* wdst0 = &wtile[tid * 32];
  __bf16* wdst1 = &wtile[128 * 32 + tid * 32];

  // Prologue: DMA W tile for k0 = 0 into buffer 0.
  async_g2l_32B(wdst0, wsrc0);
  async_g2l_32B(wdst0 + 16, wsrc0 + 16);

  // A-fragment raw f32 pipeline registers (k = k0 + hi*8 + {0..7, 16..23}).
  const float* arow = X + (size_t)(tileM + ln) * DM + hi * 8;
  __builtin_prefetch(arow, 0, 3);
  v4f a0 = *(const v4f*)(arow);
  v4f a1 = *(const v4f*)(arow + 4);
  v4f a2 = *(const v4f*)(arow + 16);
  v4f a3 = *(const v4f*)(arow + 20);

  v8f acc[8] = {};

  for (int k0 = 0; k0 < DM; k0 += 32) {
    const int buf = (k0 >> 5) & 1;

    // Convert the pipelined A chunk.
    v16bf a;
#pragma unroll
    for (int e = 0; e < 4; ++e) {
      a[e]      = (__bf16)a0[e];
      a[4 + e]  = (__bf16)a1[e];
      a[8 + e]  = (__bf16)a2[e];
      a[12 + e] = (__bf16)a3[e];
    }

    // Wait for this buffer's DMA, sync the block.
    wait_async0();
    __syncthreads();

    if (k0 + 32 < DM) {
      __bf16* nd = buf ? wdst0 : wdst1;
      const __bf16* ns = wsrc0 + k0 + 32;
      async_g2l_32B(nd, ns);
      async_g2l_32B(nd + 16, ns + 16);
      const float* ap = arow + k0 + 32;
      a0 = *(const v4f*)(ap);
      a1 = *(const v4f*)(ap + 4);
      a2 = *(const v4f*)(ap + 16);
      a3 = *(const v4f*)(ap + 20);
    }

    // 8 WMMAs; B-fragments from LDS (ds_load_b128 x2 each).
    const __bf16* wbuf = &wtile[buf * (128 * 32)];
#pragma unroll
    for (int nt = 0; nt < 8; ++nt) {
      const __bf16* bp = wbuf + (nt * 16 + ln) * 32 + hi * 8;
      v16bf bfr = join16(*(const v8bf*)(bp), *(const v8bf*)(bp + 16));
      acc[nt] = __builtin_amdgcn_wmma_f32_16x16x32_bf16(
          false, a, false, bfr, (short)0, acc[nt], false, false);
    }

    __syncthreads();   // reads done before this buffer is overwritten
  }

  // Store D: element v -> row M = tileM + v + 8*hi, col N = tileN + nt*16 + ln
#pragma unroll
  for (int nt = 0; nt < 8; ++nt) {
    const int n  = tileN + nt * 16 + ln;
    const int h  = n >> 6;
    const int dk = n & 63;
    const float bv = bias[n];
#pragma unroll
    for (int v = 0; v < 8; ++v) {
      const int row = tileM + v + 8 * hi;   // = b*S + s
      const int b   = row >> 11;
      const int s   = row & (S_ - 1);
      const size_t oidx = (((size_t)(b * H_ + h) * S_) + s) * DK + dk;
      const float val = acc[nt][v] + bv;
      bfout[oidx] = (__bf16)(val * bf_scale);
      if (q_out) q_out[oidx] = val;
    }
  }
}

// ---------------------------------------------------------------------------
// Kernel 2: flash attention.  Block = 4 waves = 4 consecutive 16-row query
// tiles of ONE (b,h).  The shared 32-key K tile (4 KB) and V tile (4 KB) are
// double-buffered in LDS via async DMA: waves 0-1 fetch K, waves 2-3 fetch V.
// Per block iteration: 32 WMMAs against 8 KB of DMA -> 4x traffic reduction.
// ---------------------------------------------------------------------------
__global__ __launch_bounds__(128) void attn_kernel(
    const __bf16* __restrict__ Qbf,   // [B,H,S,DK], pre-scaled by 1/8
    const __bf16* __restrict__ Kbf,   // [B,H,S,DK]
    const __bf16* __restrict__ Vbf,   // [B,DV,S]
    float* __restrict__ out_mean)     // [B,S,DV]
{
  __shared__ __bf16 ktile[2 * 32 * 64];   // [buf][key][d]   8 KB
  __shared__ __bf16 vtile[2 * 64 * 32];   // [buf][dv][s]    8 KB
  __shared__ float  plds[4][16 * 32];     // per-wave P tile 8 KB

  const int tid  = threadIdx.x;
  const int wave = tid >> 5;
  const int lane = tid & 31;
  const int hi   = lane >> 4;
  const int ln   = lane & 15;

  const int bh = blockIdx.x >> 5;               // b*H + h (32 blocks per bh)
  const int qt = (blockIdx.x & 31) * 4 + wave;  // 16-row tile idx within bh
  const int b  = bh >> 4;

  const __bf16* qbase = Qbf + ((size_t)bh * S_ + qt * 16) * DK;
  const __bf16* kbase = Kbf + (size_t)bh * S_ * DK;
  const __bf16* vbase = Vbf + (size_t)b * DV * S_;
  float* pl = plds[wave];

  // This thread's DMA share: waves 0-1 -> K tile, waves 2-3 -> V tile.
  const __bf16* dsrc0;   // source for s-block 0 (advance by j0*stride)
  __bf16 *ddst0, *ddst1;
  size_t dstep;          // source element step per 32-key block
  if (tid < 64) {
    const int r  = tid >> 1;          // key row 0..31
    const int co = (tid & 1) * 32;    // 64 B half of the 128 B row
    dsrc0 = kbase + (size_t)r * DK + co;
    dstep = (size_t)32 * DK;
    ddst0 = &ktile[r * 64 + co];
    ddst1 = &ktile[32 * 64 + r * 64 + co];
  } else {
    const int d = tid - 64;           // dv row 0..63, 64 B per block
    dsrc0 = vbase + (size_t)d * S_;
    dstep = 32;
    ddst0 = &vtile[d * 32];
    ddst1 = &vtile[64 * 32 + d * 32];
  }

  // Prologue: DMA K/V tiles for j0 = 0 into buffer 0.
  async_g2l_32B(ddst0, dsrc0);
  async_g2l_32B(ddst0 + 16, dsrc0 + 16);

  // Q A-fragments (loaded once).
  v16bf aq[2];
#pragma unroll
  for (int c = 0; c < 2; ++c) {
    const __bf16* qp = qbase + (size_t)ln * DK + c * 32 + hi * 8;
    aq[c] = join16(*(const v8bf*)(qp), *(const v8bf*)(qp + 16));
  }

  float rmax[8], rsum[8];
  v8f o[4] = {};
#pragma unroll
  for (int v = 0; v < 8; ++v) { rmax[v] = -3.0e38f; rsum[v] = 0.0f; }

  for (int j0 = 0; j0 < S_; j0 += 32) {
    const int buf = (j0 >> 5) & 1;

    wait_async0();
    __syncthreads();

    if (j0 + 32 < S_) {
      __bf16* nd = buf ? ddst0 : ddst1;
      const __bf16* ns = dsrc0 + (size_t)((j0 >> 5) + 1) * dstep;
      async_g2l_32B(nd, ns);
      async_g2l_32B(nd + 16, ns + 16);
    }

    // ---- scores: 16 q-rows x 32 keys, two 16x16 D tiles (K from LDS) -----
    const __bf16* kt = &ktile[buf * (32 * 64)];
    v8f sc[2];
#pragma unroll
    for (int jt = 0; jt < 2; ++jt) {
      v8f s = {};
#pragma unroll
      for (int c = 0; c < 2; ++c) {
        const __bf16* kp = kt + (jt * 16 + ln) * 64 + c * 32 + hi * 8;
        v16bf bfr = join16(*(const v8bf*)(kp), *(const v8bf*)(kp + 16));
        s = __builtin_amdgcn_wmma_f32_16x16x32_bf16(
            false, aq[c], false, bfr, (short)0, s, false, false);
      }
      sc[jt] = s;   // 1/sqrt(DK) already folded into Qbf
    }

    // ---- online softmax over the 32 columns ------------------------------
    float corr[8];
#pragma unroll
    for (int v = 0; v < 8; ++v) {
      float m0 = fmaxf(sc[0][v], sc[1][v]);
#pragma unroll
      for (int m = 1; m < 16; m <<= 1) m0 = fmaxf(m0, __shfl_xor(m0, m, 32));
      const float nm = fmaxf(rmax[v], m0);
      corr[v] = __expf(rmax[v] - nm);
      rmax[v] = nm;
      const float p0 = __expf(sc[0][v] - nm);
      const float p1 = __expf(sc[1][v] - nm);
      sc[0][v] = p0; sc[1][v] = p1;
      float dsum = p0 + p1;
#pragma unroll
      for (int m = 1; m < 16; m <<= 1) dsum += __shfl_xor(dsum, m, 32);
      rsum[v] = rsum[v] * corr[v] + dsum;
#pragma unroll
      for (int nt = 0; nt < 4; ++nt) o[nt][v] *= corr[v];
    }

    // ---- transpose P into A-fragment layout via private LDS tile ---------
#pragma unroll
    for (int v = 0; v < 8; ++v) {
      const int row = v + 8 * hi;
      pl[row * 32 + ln]      = sc[0][v];
      pl[row * 32 + 16 + ln] = sc[1][v];
    }
    v16bf apf;
    {
      const float* rp = pl + ln * 32 + hi * 8;
      v4f r0 = *(const v4f*)(rp);
      v4f r1 = *(const v4f*)(rp + 4);
      v4f r2 = *(const v4f*)(rp + 16);
      v4f r3 = *(const v4f*)(rp + 20);
#pragma unroll
      for (int e = 0; e < 4; ++e) {
        apf[e]      = (__bf16)r0[e];
        apf[4 + e]  = (__bf16)r1[e];
        apf[8 + e]  = (__bf16)r2[e];
        apf[12 + e] = (__bf16)r3[e];
      }
    }

    // ---- P (16x32) x V (32x64): V fragments from LDS ---------------------
    const __bf16* vt = &vtile[buf * (64 * 32)];
#pragma unroll
    for (int nt = 0; nt < 4; ++nt) {
      const __bf16* vp = vt + (nt * 16 + ln) * 32 + hi * 8;
      v16bf bv = join16(*(const v8bf*)(vp), *(const v8bf*)(vp + 16));
      o[nt] = __builtin_amdgcn_wmma_f32_16x16x32_bf16(
          false, apf, false, bv, (short)0, o[nt], false, false);
    }

    __syncthreads();   // reads done before this buffer is overwritten
  }

  // ---- epilogue: normalize, scale by 1/H, atomic-accumulate head mean ----
#pragma unroll
  for (int nt = 0; nt < 4; ++nt) {
#pragma unroll
    for (int v = 0; v < 8; ++v) {
      const int s = qt * 16 + v + 8 * hi;
      const float val = o[nt][v] / rsum[v] * (1.0f / H_);
      atomicAdd(&out_mean[((size_t)b * S_ + s) * DV + nt * 16 + ln], val);
    }
  }
}

// ---------------------------------------------------------------------------
extern "C" void kernel_launch(void* const* d_in, const int* in_sizes, int n_in,
                              void* d_out, int out_size, void* d_ws, size_t ws_size,
                              hipStream_t stream) {
  const float* query = (const float*)d_in[0];
  const float* key_x = (const float*)d_in[1];
  const float* value = (const float*)d_in[2];
  const float* Wq    = (const float*)d_in[3];
  const float* bq    = (const float*)d_in[4];
  const float* Wk    = (const float*)d_in[5];
  const float* bk    = (const float*)d_in[6];

  float* out      = (float*)d_out;
  float* out_mean = out;                               // [B,S,DV]
  float* out_q    = out + (size_t)B_ * S_ * DV;        // [B,H,S,DK]

  char* ws = (char*)d_ws;
  const size_t qk_bytes = (size_t)B_ * H_ * S_ * DK * sizeof(__bf16); // 16 MB
  const size_t w_bytes  = (size_t)DM * DM * sizeof(__bf16);           // 2 MB
  const size_t v_bytes  = (size_t)B_ * DV * S_ * sizeof(__bf16);      // 1 MB
  __bf16* Qbf  = (__bf16*)(ws);
  __bf16* Kbf  = (__bf16*)(ws + qk_bytes);
  __bf16* Vbf  = (__bf16*)(ws + 2 * qk_bytes);
  __bf16* Wqbf = (__bf16*)(ws + 2 * qk_bytes + v_bytes);
  __bf16* Wkbf = (__bf16*)(ws + 2 * qk_bytes + v_bytes + w_bytes);

  // Stage 0: conversions + output zeroing (memory-bound, ~17 MB total).
  const int nw8 = DM * DM / 8;
  cvt_kernel<<<(nw8 + 255) / 256, 256, 0, stream>>>(Wq, Wqbf, nw8);
  cvt_kernel<<<(nw8 + 255) / 256, 256, 0, stream>>>(Wk, Wkbf, nw8);
  const int nv8 = B_ * S_ * DV / 8;
  prep_kernel<<<(nv8 + 255) / 256, 256, 0, stream>>>(value, out_mean, Vbf, nv8);

  // Stage 1: Q/K projections (WMMA GEMMs with async-LDS W staging).
  dim3 gproj(B_ * S_ / 64, DM / 128);
  proj_kernel<<<gproj, 128, 0, stream>>>(query, Wqbf, bq, out_q, Qbf, 0.125f);
  proj_kernel<<<gproj, 128, 0, stream>>>(key_x, Wkbf, bk, nullptr, Kbf, 1.0f);

  // Stage 2: attention (WMMA flash loop with async-LDS K/V staging).
  const int ntiles = B_ * H_ * (S_ / 16);     // 8192, 4 per block
  attn_kernel<<<ntiles / 4, 128, 0, stream>>>(Qbf, Kbf, Vbf, out_mean);
}
